// ModifiedSpatialTransformer_38225208935056
// MI455X (gfx1250) — matrix operand order repaired and checked by reading, not compiled, not run on hardware
//
#include <hip/hip_runtime.h>

typedef __attribute__((ext_vector_type(16))) __bf16 v16bf;
typedef __attribute__((ext_vector_type(8)))  float  v8f;
typedef __attribute__((ext_vector_type(4)))  unsigned int v4u;
typedef __attribute__((ext_vector_type(8)))  int    v8i_t;
typedef __attribute__((ext_vector_type(4)))  int    v4i_t;

// ---------------------------------------------------------------- helpers
__device__ __forceinline__ float siluf(float x) { return x / (1.0f + __expf(-x)); }
__device__ __forceinline__ float softplusf(float x) { return (x > 20.0f) ? x : log1pf(__expf(x)); }

#define LDSW 40   // LDS row pitch in bf16 elements (80 bytes)

// Tensor Data Mover: stage a (64 rows x 32 cols) bf16 tile into LDS with
// per-row padding 64B->80B so rows land at LDSW pitch.  D# per CDNA5 ISA ch.8.
//   dim0_rem/dim1_rem: remaining tensor extent from tile origin (HW zero-fills OOB)
//   stride_elems:      row stride of the source matrix in elements
__device__ __forceinline__ void tdm_load_b_tile(const __bf16* gsrc, unsigned lds_off,
                                                long dim0_rem, long dim1_rem,
                                                long stride_elems) {
    unsigned long long ga = (unsigned long long)(uintptr_t)gsrc;
    v4u g0;
    g0[0] = 1u;                                             // count=1, user mode, no gather
    g0[1] = lds_off;                                        // lds_addr (bytes)
    g0[2] = (unsigned)(ga & 0xFFFFFFFFu);                   // global_addr[31:0]
    g0[3] = (unsigned)((ga >> 32) & 0x01FFFFFFu) | (2u << 30); // addr[56:32] | type=2
    unsigned d0 = (unsigned)(dim0_rem > 0 ? dim0_rem : 0);
    unsigned d1 = (unsigned)(dim1_rem > 0 ? dim1_rem : 0);
    unsigned long long st = (unsigned long long)stride_elems;
    v8i_t g1;
    // data_size=1(2B) | pad_enable | pad_interval=3(16 DW) | pad_amount=3(4 DW)
    g1[0] = (int)((1u << 16) | (1u << 20) | (3u << 22) | (3u << 25));
    g1[1] = (int)((d0 & 0xFFFFu) << 16);                    // tensor_dim0[15:0] @ bit48
    g1[2] = (int)(((d0 >> 16) & 0xFFFFu) | ((d1 & 0xFFFFu) << 16));
    g1[3] = (int)(((d1 >> 16) & 0xFFFFu) | (32u << 16));    // tile_dim0 = 32
    g1[4] = 64;                                             // tile_dim1 = 64, tile_dim2 = 0
    g1[5] = (int)(unsigned)(st & 0xFFFFFFFFu);              // tensor_dim0_stride lo
    g1[6] = (int)(unsigned)((st >> 32) & 0xFFFFu);          // stride hi | dim1_stride=0
    g1[7] = 0;
    v4i_t z4 = {0, 0, 0, 0};                                // groups 2/3 unused (2D)
    v8i_t z8 = {0, 0, 0, 0, 0, 0, 0, 0};
    __builtin_amdgcn_tensor_load_to_lds(g0, g1, z4, z4, z8, 0);
}

// ---------------------------------------------------------------- packing
__global__ void pack_bf16_kernel(const float* __restrict__ src, __bf16* __restrict__ dst, int n) {
    int i = blockIdx.x * 256 + threadIdx.x;
    if (i < n) dst[i] = (__bf16)src[i];
}

// conv weight (O=256, I=256, 3, 3) -> (O, K=2304), k = (ky*3+kx)*256 + cin
__global__ void convrepack_kernel(const float* __restrict__ w, __bf16* __restrict__ dst) {
    int idx = blockIdx.x * 256 + threadIdx.x;           // 256*2304
    if (idx >= 256 * 2304) return;
    int k = idx % 2304, n = idx / 2304;
    int pos = k / 256, cin = k % 256;
    int ky = pos / 3, kx = pos % 3;
    dst[idx] = (__bf16)w[((n * 256 + cin) * 3 + ky) * 3 + kx];
}

// BN folding: scale = g*rsqrt(v+eps), shift = b - m*scale
__global__ void bnprep_kernel(const float* __restrict__ g, const float* __restrict__ b,
                              const float* __restrict__ m, const float* __restrict__ v,
                              float* __restrict__ scale, float* __restrict__ shift, int n) {
    int i = blockIdx.x * 256 + threadIdx.x;
    if (i < n) {
        float sc = g[i] * rsqrtf(v[i] + 1e-5f);
        scale[i] = sc;
        shift[i] = b[i] - m[i] * sc;
    }
}

// ---------------------------------------------------------------- normalizations
// GroupNorm (32 groups, C=1024, 32x32 spatial) -> channel-last (B, L=1024, C=1024)
__global__ __launch_bounds__(256)
void groupnorm_cl_kernel(const float* __restrict__ x, const float* __restrict__ g,
                         const float* __restrict__ bt, float* __restrict__ out) {
    __shared__ float s_sum[256], s_sq[256];
    int bb = blockIdx.x >> 5;
    int grp = blockIdx.x & 31;
    int ch0 = grp * 32;
    const int NE = 32 * 1024;
    size_t xbase = ((size_t)bb * 1024 + ch0) * 1024;   // group channels are contiguous
    float s = 0.f, ss = 0.f;
    for (int i = threadIdx.x; i < NE; i += 256) {
        float v = x[xbase + i];
        s += v; ss += v * v;
    }
    s_sum[threadIdx.x] = s; s_sq[threadIdx.x] = ss;
    __syncthreads();
    for (int off = 128; off > 0; off >>= 1) {
        if (threadIdx.x < off) { s_sum[threadIdx.x] += s_sum[threadIdx.x + off];
                                 s_sq[threadIdx.x]  += s_sq[threadIdx.x + off]; }
        __syncthreads();
    }
    float mu  = s_sum[0] / (float)NE;
    float var = s_sq[0] / (float)NE - mu * mu;
    float rs  = rsqrtf(var + 1e-6f);
    for (int i = threadIdx.x; i < NE; i += 256) {
        int c = i & 31, p = i >> 5;
        int ch = ch0 + c;
        float v = x[((size_t)bb * 1024 + ch) * 1024 + p];
        out[((size_t)bb * 1024 + p) * 1024 + ch] = (v - mu) * rs * g[ch] + bt[ch];
    }
}

// (B, C=1024, P=1024) -> (B, P, C) tiled transpose
__global__ __launch_bounds__(256)
void transpose1024_kernel(const float* __restrict__ in, float* __restrict__ out) {
    __shared__ float tile[32][33];
    int b = blockIdx.z;
    int c0 = blockIdx.x * 32, p0 = blockIdx.y * 32;
    int tx = threadIdx.x & 31, ty = threadIdx.x >> 5;   // 32 x 8
    for (int j = 0; j < 32; j += 8)
        tile[ty + j][tx] = in[((size_t)b * 1024 + c0 + ty + j) * 1024 + p0 + tx];
    __syncthreads();
    for (int j = 0; j < 32; j += 8)
        out[((size_t)b * 1024 + p0 + ty + j) * 1024 + c0 + tx] = tile[tx][ty + j];
}

// row-wise LayerNorm over C, rows = 2048
__global__ __launch_bounds__(256)
void layernorm_kernel(const float* __restrict__ in, int C,
                      const float* __restrict__ g, const float* __restrict__ b,
                      float* __restrict__ out) {
    __shared__ float s_sum[256], s_sq[256];
    size_t row = blockIdx.x;
    const float* p = in + row * C;
    float s = 0.f, ss = 0.f;
    for (int i = threadIdx.x; i < C; i += 256) { float v = p[i]; s += v; ss += v * v; }
    s_sum[threadIdx.x] = s; s_sq[threadIdx.x] = ss;
    __syncthreads();
    for (int off = 128; off > 0; off >>= 1) {
        if (threadIdx.x < off) { s_sum[threadIdx.x] += s_sum[threadIdx.x + off];
                                 s_sq[threadIdx.x]  += s_sq[threadIdx.x + off]; }
        __syncthreads();
    }
    float mu  = s_sum[0] / (float)C;
    float var = s_sq[0] / (float)C - mu * mu;
    float rs  = rsqrtf(var + 1e-5f);
    for (int i = threadIdx.x; i < C; i += 256)
        out[row * C + i] = (p[i] - mu) * rs * g[i] + b[i];
}

// ---------------------------------------------------------------- WMMA GEMM
// C[M,N] = act( (A[M,K] @ W[N,K]^T) * scale[n] + shift[n] (+bias[n]) (+addsrc[m,n]) )
// A fp32 row-major (lda), W bf16 row-major NxK.  Block: 128 thr = 4 waves,
// tile 64x64, BK=32, wave w computes rows [w*16, w*16+16) x 64 cols (4 WMMA accs).
// tdm!=0: weight tile staged by the Tensor Data Mover (wave 0 issues, per-row
// padding lands rows at the LDSW pitch); A tile converted fp32->bf16 by VALU in
// parallel with the DMA.
__global__ __launch_bounds__(128)
void gemm_bf16_kernel(const float* __restrict__ A, int lda,
                      const __bf16* __restrict__ W,
                      float* __restrict__ C, int ldc,
                      int M, int N, int K,
                      const float* __restrict__ bias,
                      const float* __restrict__ scale,
                      const float* __restrict__ shift,
                      const float* __restrict__ addsrc, int ldadd,
                      int act, int tdm) {
    __shared__ __bf16 As[64 * LDSW];
    __shared__ __bf16 Bs[64 * LDSW];
    int tid = threadIdx.x, lane = tid & 31, wave = tid >> 5;
    int m0 = blockIdx.y * 64;
    int n0 = blockIdx.x * 64;
    int lrow = tid >> 1;              // 0..63
    int lcol = (tid & 1) * 16;        // 0 or 16
    int gm = m0 + lrow;
    int gn = n0 + lrow;
    const float*  aptr = (gm < M) ? (A + (size_t)gm * lda + lcol) : nullptr;
    const __bf16* bptr = (gn < N) ? (W + (size_t)gn * K + lcol) : nullptr;
    unsigned bs_lds = (unsigned)(uintptr_t)(void*)Bs;
    v8f acc[4] = {};
    for (int k0 = 0; k0 < K; k0 += 32) {
        if (tdm) {
            if (wave == 0) {
                tdm_load_b_tile(W + (size_t)n0 * K + k0, bs_lds,
                                (long)K - k0, (long)N - n0, (long)K);
            }
        } else {   // VALU staging of B
            __bf16* dst = Bs + lrow * LDSW + lcol;
            if (bptr) {
                const __bf16* s = bptr + k0;
                if (k0 + 32 < K) __builtin_prefetch(s + 32, 0, 1);
#pragma unroll
                for (int j = 0; j < 16; ++j) dst[j] = s[j];
            } else {
#pragma unroll
                for (int j = 0; j < 16; ++j) dst[j] = (__bf16)0.0f;
            }
        }
        {   // stage A (fp32 -> bf16), overlaps the TDM transfer
            __bf16* dst = As + lrow * LDSW + lcol;
            if (aptr) {
                const float* s = aptr + k0;
                if (k0 + 32 < K) __builtin_prefetch(s + 32, 0, 1);
#pragma unroll
                for (int j = 0; j < 16; ++j) dst[j] = (__bf16)s[j];
            } else {
#pragma unroll
                for (int j = 0; j < 16; ++j) dst[j] = (__bf16)0.0f;
            }
        }
        if (tdm && wave == 0) __builtin_amdgcn_s_wait_tensorcnt(0);
        __syncthreads();
        // A fragment (ISA 16-bit 16x32 layout): lane<16 -> K 0..7 / 16..23, lane>=16 -> +8
        int half = lane >> 4;
        const __bf16* pa = As + (wave * 16 + (lane & 15)) * LDSW;
        v16bf a;
#pragma unroll
        for (int j = 0; j < 8; ++j) a[j] = pa[half * 8 + j];
#pragma unroll
        for (int j = 0; j < 8; ++j) a[8 + j] = pa[16 + half * 8 + j];
#pragma unroll
        for (int t = 0; t < 4; ++t) {
            // B fragment: lane%16 = N col, lane<16 -> K 0..7 / 16..23, lane>=16 -> +8
            const __bf16* pb = Bs + (t * 16 + (lane & 15)) * LDSW + half * 8;
            v16bf bfr;
#pragma unroll
            for (int j = 0; j < 16; ++j) bfr[j] = pb[(j < 8) ? j : j + 8];
            acc[t] = __builtin_amdgcn_wmma_f32_16x16x32_bf16(
                false, a, false, bfr, (short)0, acc[t], false, false);
        }
        __syncthreads();
    }
    // epilogue: C 16x16 layout: VGPR r -> M = r (+8 for lanes 16..31), N = lane%16
    int mlocal = (lane < 16) ? 0 : 8;
    int nlane = lane & 15;
    for (int t = 0; t < 4; ++t) {
        int n = n0 + t * 16 + nlane;
        if (n >= N) continue;
        float sc = scale ? scale[n] : 1.0f;
        float sh = shift ? shift[n] : (bias ? bias[n] : 0.0f);
#pragma unroll
        for (int r = 0; r < 8; ++r) {
            int m = m0 + wave * 16 + mlocal + r;
            if (m >= M) continue;
            float v = acc[t][r] * sc + sh;
            if (addsrc) v += addsrc[(size_t)m * ldadd + n];
            if (act == 1) v = fmaxf(v, 0.0f);
            else if (act == 2) v = softplusf(v);
            C[(size_t)m * ldc + n] = v;
        }
    }
}

// ---------------------------------------------------------------- mamba pieces
// causal depthwise conv (D_CONV=4) + silu; xz (B,L,2*d_in) channel-last, xc = first d_in
__global__ void causalconv_kernel(const float* __restrict__ xz, const float* __restrict__ w,
                                  const float* __restrict__ b, float* __restrict__ u,
                                  int d_in, int L, long total) {
    long idx = (long)blockIdx.x * 256 + threadIdx.x;
    if (idx >= total) return;
    int c = (int)(idx % d_in);
    long row = idx / d_in;          // b*L + l
    int l = (int)(row % L);
    float acc = b[c];
#pragma unroll
    for (int k = 0; k < 4; ++k) {
        int ll = l - 3 + k;
        if (ll >= 0) acc += w[c * 4 + k] * xz[(row - l + ll) * (size_t)(2 * d_in) + c];
    }
    u[idx] = siluf(acc);
}

// tiny-rank dt projection (mamba1, R=4) + softplus
__global__ void dt_small_kernel(const float* __restrict__ dbc, int rs,
                                const float* __restrict__ dtw, const float* __restrict__ dtb,
                                float* __restrict__ dt, int d_in, int R, long total) {
    long idx = (long)blockIdx.x * 256 + threadIdx.x;
    if (idx >= total) return;
    int c = (int)(idx % d_in);
    long row = idx / d_in;
    float acc = dtb[c];
    for (int r = 0; r < R; ++r) acc += dbc[row * rs + r] * dtw[c * R + r];
    dt[idx] = softplusf(acc);
}

// selective scan over L; one thread per channel, 16 states in registers.
// fused tail: y = (scan + u*D[c]) * silu(z)
__global__ __launch_bounds__(128)
void scan_kernel(const float* __restrict__ dt, const float* __restrict__ u,
                 const float* __restrict__ dbc, int R,
                 const float* __restrict__ xz, int d_in,
                 const float* __restrict__ A_log, const float* __restrict__ Dp,
                 float* __restrict__ y, int L) {
    int c = blockIdx.x * 128 + threadIdx.x;
    int b = blockIdx.y;
    float h[16];
    float Ar[16];
#pragma unroll
    for (int n = 0; n < 16; ++n) { h[n] = 0.0f; Ar[n] = -__expf(A_log[c * 16 + n]); }
    float Dc = Dp[c];
    __shared__ float BC[32];
    const int rs = R + 32;
    for (int l = 0; l < L; ++l) {
        size_t row = (size_t)b * L + l;
        if (threadIdx.x < 32) BC[threadIdx.x] = dbc[row * rs + R + threadIdx.x];
        __syncthreads();
        float dtv = dt[row * d_in + c];
        float uv  = u[row * d_in + c];
        float du  = dtv * uv;
        float yacc = 0.0f;
#pragma unroll
        for (int n = 0; n < 16; ++n) {
            h[n] = __expf(dtv * Ar[n]) * h[n] + du * BC[n];
            yacc += h[n] * BC[16 + n];
        }
        float zv = xz[row * (size_t)(2 * d_in) + d_in + c];
        y[row * d_in + c] = (yacc + uv * Dc) * siluf(zv);
        __syncthreads();
    }
}

// ---------------------------------------------------------------- shuffle + conv3x3
// h channel-last (B,1024,1024ch) -> shuffled channel-last (B, 64*64, 256)
__global__ void pixshuf_kernel(const float* __restrict__ hin, float* __restrict__ outcl) {
    int idx = blockIdx.x * 256 + threadIdx.x;     // 2*64*64*256
    int c = idx & 255; int rest = idx >> 8;
    int x2 = rest & 63; rest >>= 6;
    int y2 = rest & 63; int b = rest >> 6;
    int ch = 4 * c + 2 * (x2 & 1) + (y2 & 1);
    int p = (y2 >> 1) * 32 + (x2 >> 1);
    outcl[idx] = hin[((size_t)b * 1024 + p) * 1024 + ch];
}

// implicit-GEMM 3x3 conv: X channel-last (B, 64*64, 256), Wp (256, 2304) bf16,
// out (B,256,64,64) fp32.  One block = one output row y, 64 pixels x 64 out-channels.
// Weight tiles staged by the TDM; activation tiles converted by VALU in parallel.
__global__ __launch_bounds__(128)
void conv3x3_wmma_kernel(const float* __restrict__ X, const __bf16* __restrict__ Wp,
                         const float* __restrict__ bias, float* __restrict__ out) {
    __shared__ __bf16 As[64 * LDSW];
    __shared__ __bf16 Bs[64 * LDSW];
    int b = blockIdx.z;
    int y = blockIdx.y;
    int n0 = blockIdx.x * 64;
    int tid = threadIdx.x, lane = tid & 31, wave = tid >> 5;
    int lrow = tid >> 1;
    int lcol = (tid & 1) * 16;
    unsigned bs_lds = (unsigned)(uintptr_t)(void*)Bs;
    v8f acc[4] = {};
    for (int pos = 0; pos < 9; ++pos) {
        int ky = pos / 3, kx = pos % 3;
        int yy = y + ky - 1;
        if (yy < 0 || yy >= 64) continue;          // uniform across block
        int xx = lrow + kx - 1;
        bool valid = (xx >= 0 && xx < 64);
        const float* srcA = X + (((size_t)b * 4096) + (size_t)yy * 64 + xx) * 256;
        for (int c0 = 0; c0 < 256; c0 += 32) {
            int kk = pos * 256 + c0;
            if (wave == 0) {
                tdm_load_b_tile(Wp + (size_t)n0 * 2304 + kk, bs_lds,
                                (long)2304 - kk, (long)256 - n0, 2304L);
            }
            {
                __bf16* dst = As + lrow * LDSW + lcol;
                if (valid) {
                    const float* s = srcA + c0 + lcol;
#pragma unroll
                    for (int j = 0; j < 16; ++j) dst[j] = (__bf16)s[j];
                } else {
#pragma unroll
                    for (int j = 0; j < 16; ++j) dst[j] = (__bf16)0.0f;
                }
            }
            if (wave == 0) __builtin_amdgcn_s_wait_tensorcnt(0);
            __syncthreads();
            int half = lane >> 4;
            const __bf16* pa = As + (wave * 16 + (lane & 15)) * LDSW;
            v16bf a;
#pragma unroll
            for (int j = 0; j < 8; ++j) a[j] = pa[half * 8 + j];
#pragma unroll
            for (int j = 0; j < 8; ++j) a[8 + j] = pa[16 + half * 8 + j];
#pragma unroll
            for (int t = 0; t < 4; ++t) {
                const __bf16* pb = Bs + (t * 16 + (lane & 15)) * LDSW + half * 8;
                v16bf bfr;
#pragma unroll
                for (int j = 0; j < 16; ++j) bfr[j] = pb[(j < 8) ? j : j + 8];
                acc[t] = __builtin_amdgcn_wmma_f32_16x16x32_bf16(
                    false, a, false, bfr, (short)0, acc[t], false, false);
            }
            __syncthreads();
        }
    }
    int mlocal = (lane < 16) ? 0 : 8;
    int nlane = lane & 15;
#pragma unroll
    for (int t = 0; t < 4; ++t) {
        int n = n0 + t * 16 + nlane;
        float bv = bias[n];
#pragma unroll
        for (int r = 0; r < 8; ++r) {
            int x = wave * 16 + mlocal + r;
            out[(((size_t)b * 256 + n) * 64 + y) * 64 + x] = acc[t][r] + bv;
        }
    }
}

// ---------------------------------------------------------------- host side
static void launch_gemm(hipStream_t s, const float* A, int lda, const __bf16* W,
                        float* C, int ldc, int M, int N, int K,
                        const float* bias, const float* scale, const float* shift,
                        const float* addsrc, int ldadd, int act, int tdm) {
    dim3 g((N + 63) / 64, (M + 63) / 64);
    gemm_bf16_kernel<<<g, 128, 0, s>>>(A, lda, W, C, ldc, M, N, K,
                                       bias, scale, shift, addsrc, ldadd, act, tdm);
}

struct MambaScratch { float *tn, *xz, *u, *dbc, *dt, *y; };

static void run_mamba(hipStream_t s, const float* in_cl, int d, int d_in, int R,
                      const float* ln_g, const float* ln_b,
                      const __bf16* inw_bf, const float* convw, const float* convb,
                      const __bf16* xpw_bf, const float* dtw_f32, const __bf16* dtw_bf,
                      const float* dtb, const float* A_log, const float* Dp,
                      const __bf16* outw_bf, float* out, const float* addsrc,
                      const MambaScratch& ws) {
    const int Mrows = 2048, L = 1024, Bb = 2;
    layernorm_kernel<<<Mrows, 256, 0, s>>>(in_cl, d, ln_g, ln_b, ws.tn);
    launch_gemm(s, ws.tn, d, inw_bf, ws.xz, 2 * d_in, Mrows, 2 * d_in, d,
                nullptr, nullptr, nullptr, nullptr, 0, 0, 1);
    long nconv = (long)Mrows * d_in;
    causalconv_kernel<<<(unsigned)((nconv + 255) / 256), 256, 0, s>>>(
        ws.xz, convw, convb, ws.u, d_in, L, nconv);
    launch_gemm(s, ws.u, d_in, xpw_bf, ws.dbc, R + 32, Mrows, R + 32, d_in,
                nullptr, nullptr, nullptr, nullptr, 0, 0, /*ragged N -> VALU*/0);
    if (R == 4) {
        dt_small_kernel<<<(unsigned)((nconv + 255) / 256), 256, 0, s>>>(
            ws.dbc, R + 32, dtw_f32, dtb, ws.dt, d_in, R, nconv);
    } else {
        launch_gemm(s, ws.dbc, R + 32, dtw_bf, ws.dt, d_in, Mrows, d_in, R,
                    dtb, nullptr, nullptr, nullptr, 0, /*softplus*/2, 1);
    }
    dim3 sg(d_in / 128, Bb);
    scan_kernel<<<sg, 128, 0, s>>>(ws.dt, ws.u, ws.dbc, R, ws.xz, d_in, A_log, Dp, ws.y, L);
    launch_gemm(s, ws.y, d_in, outw_bf, out, d, Mrows, d, d_in,
                nullptr, nullptr, nullptr, addsrc, d, 0, 1);
}

enum {
    IN_X = 0, IN_CTX,
    GN_G, GN_B,
    IR1_W1, IR1_BN1_G, IR1_BN1_B, IR1_BN1_M, IR1_BN1_V,
    IR1_W2, IR1_BN2_G, IR1_BN2_B, IR1_BN2_M, IR1_BN2_V,
    IR1_W3, IR1_BN3_G, IR1_BN3_B, IR1_BN3_M, IR1_BN3_V,
    IR1_W4,
    PCTX_W, PCTX_B,
    IR2_W1, IR2_BN1_G, IR2_BN1_B, IR2_BN1_M, IR2_BN1_V,
    IR2_W2, IR2_BN2_G, IR2_BN2_B, IR2_BN2_M, IR2_BN2_V,
    IR2_W3, IR2_BN3_G, IR2_BN3_B, IR2_BN3_M, IR2_BN3_V,
    IR2_W4,
    CONV_W, CONV_B,
    M1_LNG, M1_LNB, M1_INW, M1_CONVW, M1_CONVB, M1_XPW, M1_DTW, M1_DTB, M1_ALOG, M1_D, M1_OUTW,
    M2_LNG, M2_LNB, M2_INW, M2_CONVW, M2_CONVB, M2_XPW, M2_DTW, M2_DTB, M2_ALOG, M2_D, M2_OUTW,
    N_INPUTS
};

extern "C" void kernel_launch(void* const* d_in, const int* in_sizes, int n_in,
                              void* d_out, int out_size, void* d_ws, size_t ws_size,
                              hipStream_t stream) {
    (void)in_sizes; (void)n_in; (void)out_size; (void)ws_size;
    const float* P[N_INPUTS];
    for (int j = 0; j < N_INPUTS; ++j) P[j] = (const float*)d_in[j];

    // ---- workspace bump allocator
    char* wp = (char*)d_ws;
    auto allocF = [&](size_t elems) -> float* {
        float* p = (float*)wp; wp += ((elems * 4 + 255) & ~size_t(255)); return p; };
    auto allocH = [&](size_t elems) -> __bf16* {
        __bf16* p = (__bf16*)wp; wp += ((elems * 2 + 255) & ~size_t(255)); return p; };

    const int Mrows = 2048;
    float* act0  = allocF((size_t)Mrows * 1024);   // GN output, channel-last; reused for g2
    float* ctxcl = allocF((size_t)Mrows * 1024);   // context channel-last; reused for m2out
    float* h1    = allocF((size_t)Mrows * 512);    // reused for shuffled_cl
    float* h2    = allocF((size_t)Mrows * 128);
    float* g1    = allocF((size_t)Mrows * 64);
    float* gctx  = allocF((size_t)Mrows * 64);
    float* mh    = allocF((size_t)Mrows * 64);
    float* fused = allocF((size_t)Mrows * 64);
    float* o1    = allocF((size_t)Mrows * 256);
    float* o2    = allocF((size_t)Mrows * 512);
    MambaScratch mw;
    mw.tn  = allocF((size_t)Mrows * 1024);
    mw.xz  = allocF((size_t)Mrows * 4096);
    mw.u   = allocF((size_t)Mrows * 2048);
    mw.dbc = allocF((size_t)Mrows * 96);
    mw.dt  = allocF((size_t)Mrows * 2048);
    mw.y   = allocF((size_t)Mrows * 2048);
    float* g2    = act0;     // alias (act0 dead after ir_in)
    float* m2out = ctxcl;    // alias (ctxcl dead after proj_ctx)
    float* shuf  = h1;       // alias (h1 dead after ir_in)

    // BN folded params
    float *sc1i = allocF(512), *sh1i = allocF(512);
    float *sc2i = allocF(128), *sh2i = allocF(128);
    float *sc3i = allocF(64),  *sh3i = allocF(64);
    float *sc1o = allocF(256), *sh1o = allocF(256);
    float *sc2o = allocF(512), *sh2o = allocF(512);
    float *sc3o = allocF(1024), *sh3o = allocF(1024);

    // bf16 weight packs
    __bf16* w1i  = allocH(512 * 1024);
    __bf16* w2i  = allocH(128 * 512);
    __bf16* w3i  = allocH(64 * 128);
    __bf16* w4i  = allocH(64 * 1024);
    __bf16* pcw  = allocH(64 * 1024);
    __bf16* w1o  = allocH(256 * 64);
    __bf16* w2o  = allocH(512 * 256);
    __bf16* w3o  = allocH(1024 * 512);
    __bf16* w4o  = allocH(1024 * 64);
    __bf16* inw1 = allocH(256 * 64);
    __bf16* xpw1 = allocH(36 * 128);
    __bf16* ow1  = allocH(64 * 128);
    __bf16* inw2 = allocH(4096 * 1024);
    __bf16* xpw2 = allocH(96 * 2048);
    __bf16* dtw2 = allocH(2048 * 64);
    __bf16* ow2  = allocH(1024 * 2048);
    __bf16* cwp  = allocH(256 * 2304);

    auto pack = [&](const float* src, __bf16* dst, int n) {
        pack_bf16_kernel<<<(n + 255) / 256, 256, 0, stream>>>(src, dst, n);
    };
    pack(P[IR1_W1], w1i, 512 * 1024);  pack(P[IR1_W2], w2i, 128 * 512);
    pack(P[IR1_W3], w3i, 64 * 128);    pack(P[IR1_W4], w4i, 64 * 1024);
    pack(P[PCTX_W], pcw, 64 * 1024);
    pack(P[IR2_W1], w1o, 256 * 64);    pack(P[IR2_W2], w2o, 512 * 256);
    pack(P[IR2_W3], w3o, 1024 * 512);  pack(P[IR2_W4], w4o, 1024 * 64);
    pack(P[M1_INW], inw1, 256 * 64);   pack(P[M1_XPW], xpw1, 36 * 128);
    pack(P[M1_OUTW], ow1, 64 * 128);
    pack(P[M2_INW], inw2, 4096 * 1024); pack(P[M2_XPW], xpw2, 96 * 2048);
    pack(P[M2_DTW], dtw2, 2048 * 64);   pack(P[M2_OUTW], ow2, 1024 * 2048);
    convrepack_kernel<<<(256 * 2304 + 255) / 256, 256, 0, stream>>>(P[CONV_W], cwp);

    auto bnp = [&](int gi, float* sc, float* sh, int n) {
        bnprep_kernel<<<(n + 255) / 256, 256, 0, stream>>>(
            P[gi], P[gi + 1], P[gi + 2], P[gi + 3], sc, sh, n);
    };
    bnp(IR1_BN1_G, sc1i, sh1i, 512); bnp(IR1_BN2_G, sc2i, sh2i, 128);
    bnp(IR1_BN3_G, sc3i, sh3i, 64);
    bnp(IR2_BN1_G, sc1o, sh1o, 256); bnp(IR2_BN2_G, sc2o, sh2o, 512);
    bnp(IR2_BN3_G, sc3o, sh3o, 1024);

    // ---- pipeline
    // 1. GroupNorm -> channel-last
    groupnorm_cl_kernel<<<64, 256, 0, stream>>>(P[IN_X], P[GN_G], P[GN_B], act0);
    // 2. ir_in (GT block), BN/ReLU fused into GEMM epilogues
    launch_gemm(stream, act0, 1024, w1i, h1, 512, Mrows, 512, 1024,
                nullptr, sc1i, sh1i, nullptr, 0, 1, 1);
    launch_gemm(stream, h1, 512, w2i, h2, 128, Mrows, 128, 512,
                nullptr, sc2i, sh2i, nullptr, 0, 1, 1);
    launch_gemm(stream, h2, 128, w3i, g1, 64, Mrows, 64, 128,
                nullptr, sc3i, sh3i, nullptr, 0, 0, 1);
    launch_gemm(stream, act0, 1024, w4i, g1, 64, Mrows, 64, 1024,
                nullptr, sc3i, sh3i, g1, 64, 0, 1);      // skip path (+bn3, faithful)
    // 3. context: transpose -> 1x1 proj (+bias)
    {
        dim3 tg(32, 32, 2);
        transpose1024_kernel<<<tg, 256, 0, stream>>>(P[IN_CTX], ctxcl);
    }
    launch_gemm(stream, ctxcl, 1024, pcw, gctx, 64, Mrows, 64, 1024,
                P[PCTX_B], nullptr, nullptr, nullptr, 0, 0, 1);
    // 4. mamba1 on h, then on ctx with fused residual add
    run_mamba(stream, g1, 64, 128, 4, P[M1_LNG], P[M1_LNB], inw1,
              P[M1_CONVW], P[M1_CONVB], xpw1, P[M1_DTW], nullptr, P[M1_DTB],
              P[M1_ALOG], P[M1_D], ow1, mh, nullptr, mw);
    run_mamba(stream, gctx, 64, 128, 4, P[M1_LNG], P[M1_LNB], inw1,
              P[M1_CONVW], P[M1_CONVB], xpw1, P[M1_DTW], nullptr, P[M1_DTB],
              P[M1_ALOG], P[M1_D], ow1, fused, mh, mw);
    // 5. ir_out
    launch_gemm(stream, fused, 64, w1o, o1, 256, Mrows, 256, 64,
                nullptr, sc1o, sh1o, nullptr, 0, 1, 1);
    launch_gemm(stream, o1, 256, w2o, o2, 512, Mrows, 512, 256,
                nullptr, sc2o, sh2o, nullptr, 0, 1, 1);
    launch_gemm(stream, o2, 512, w3o, g2, 1024, Mrows, 1024, 512,
                nullptr, sc3o, sh3o, nullptr, 0, 0, 1);
    launch_gemm(stream, fused, 64, w4o, g2, 1024, Mrows, 1024, 64,
                nullptr, sc3o, sh3o, g2, 1024, 0, 1);
    // 6. mamba2
    run_mamba(stream, g2, 1024, 2048, 64, P[M2_LNG], P[M2_LNB], inw2,
              P[M2_CONVW], P[M2_CONVB], xpw2, nullptr, dtw2, P[M2_DTB],
              P[M2_ALOG], P[M2_D], ow2, m2out, nullptr, mw);
    // 7. pixel shuffle -> channel-last (B,64*64,256)
    pixshuf_kernel<<<(2 * 64 * 64 * 256) / 256, 256, 0, stream>>>(m2out, shuf);
    // 8. 3x3 conv (implicit GEMM, WMMA) -> d_out (B,256,64,64)
    {
        dim3 cg(4, 64, 2);
        conv3x3_wmma_kernel<<<cg, 128, 0, stream>>>(shuf, cwp, P[CONV_B], (float*)d_out);
    }
}
